// WordAttention_12953621365291
// MI455X (gfx1250) — compile-verified, matrix operand order checked by reading, not run on hardware
//
#include <hip/hip_runtime.h>
#include <hip/hip_bf16.h>

typedef __attribute__((ext_vector_type(16))) __bf16 v16bf;
typedef __attribute__((ext_vector_type(8)))  __bf16 v8bf;
typedef __attribute__((ext_vector_type(8)))  float  v8f;

#define Bn 8
#define Sn 2048
#define Dn 512
#define Hn 512

// Build a 16-element bf16 fragment from two contiguous 8-element (16B) chunks.
// ISA 16-bit A/B layout: per lane, K = {kb..kb+7} and {kb+16..kb+23},
// kb = (lane<16 ? 0 : 8); caller bakes kb into the pointers.
__device__ __forceinline__ v16bf make_frag(const __bf16* p0, const __bf16* p1) {
    v8bf a0 = *(const v8bf*)p0;
    v8bf a1 = *(const v8bf*)p1;
    v16bf r;
#pragma unroll
    for (int i = 0; i < 8; ++i) { r[i] = a0[i]; r[i + 8] = a1[i]; }
    return r;
}

__device__ __forceinline__ v8f wmma_bf16(v16bf a, v16bf b, v8f c) {
    return __builtin_amdgcn_wmma_f32_16x16x32_bf16(
        /*neg_a=*/false, a, /*neg_b=*/false, b,
        /*c_mod=*/(short)0, c, /*reuse_a=*/false, /*reuse_b=*/false);
}

// ---------------- conversion kernels ----------------

__global__ __launch_bounds__(256) void cvt_x_kernel(const float* __restrict__ x,
                                                    __bf16* __restrict__ xb, int n) {
    for (int i = blockIdx.x * blockDim.x + threadIdx.x; i < n; i += gridDim.x * blockDim.x)
        xb[i] = (__bf16)x[i];
}

// Wt[w][h][d] = W_w[d][h]  (transpose so WMMA B-frag loads are contiguous in d)
__global__ __launch_bounds__(256) void cvt_w_kernel(const float* __restrict__ Wq,
                                                    const float* __restrict__ Wk,
                                                    const float* __restrict__ Wv,
                                                    __bf16* __restrict__ Wt) {
    int idx = blockIdx.x * blockDim.x + threadIdx.x;
    if (idx >= 3 * Dn * Hn) return;
    int w   = idx >> 18;            // / (512*512)
    int rem = idx & (Dn * Hn - 1);
    int h   = rem >> 9;             // / 512
    int d   = rem & (Dn - 1);
    const float* W = (w == 0) ? Wq : ((w == 1) ? Wk : Wv);
    Wt[idx] = (__bf16)W[(size_t)d * Hn + h];
}

// ---------------- fused QKV projection ----------------
// One wave per 16x64 output tile (4 WMMA accumulators, A-frag reused 4x).
// mat: 0=Q, 1=K (row-major [B*S,H] bf16), 2=V stored transposed Vt[B][H][S].
__global__ __launch_bounds__(128) void qkv_kernel(const __bf16* __restrict__ xb,
                                                  const __bf16* __restrict__ Wt,
                                                  const float* __restrict__ bq,
                                                  const float* __restrict__ bk,
                                                  const float* __restrict__ bv,
                                                  __bf16* __restrict__ Qb,
                                                  __bf16* __restrict__ Kb,
                                                  __bf16* __restrict__ Vt) {
    const int lane = threadIdx.x & 31;
    const int wid  = blockIdx.x * 4 + (threadIdx.x >> 5);   // 3*1024*8 = 24576 waves
    const int mat     = wid >> 13;                          // / (1024*8)
    const int rem     = wid & 8191;
    const int rowTile = rem >> 3;                           // B*S/16 = 1024 row tiles
    const int col0    = (rem & 7) << 6;                     // H/64  = 8 col groups
    if (mat >= 3) return;

    const int m  = lane & 15;
    const int kb = (lane < 16) ? 0 : 8;

    const __bf16* xrow = xb + (size_t)(rowTile * 16 + m) * Dn;
    const __bf16* wbase = Wt + (size_t)mat * Dn * Hn;
    const __bf16* wrow0 = wbase + (size_t)(col0 +  0 + m) * Dn;
    const __bf16* wrow1 = wbase + (size_t)(col0 + 16 + m) * Dn;
    const __bf16* wrow2 = wbase + (size_t)(col0 + 32 + m) * Dn;
    const __bf16* wrow3 = wbase + (size_t)(col0 + 48 + m) * Dn;

    v8f acc0 = {}, acc1 = {}, acc2 = {}, acc3 = {};
#pragma unroll 2
    for (int k = 0; k < Dn; k += 32) {
        v16bf a  = make_frag(xrow  + k + kb, xrow  + k + 16 + kb);
        v16bf b0 = make_frag(wrow0 + k + kb, wrow0 + k + 16 + kb);
        v16bf b1 = make_frag(wrow1 + k + kb, wrow1 + k + 16 + kb);
        v16bf b2 = make_frag(wrow2 + k + kb, wrow2 + k + 16 + kb);
        v16bf b3 = make_frag(wrow3 + k + kb, wrow3 + k + 16 + kb);
        acc0 = wmma_bf16(a, b0, acc0);
        acc1 = wmma_bf16(a, b1, acc1);
        acc2 = wmma_bf16(a, b2, acc2);
        acc3 = wmma_bf16(a, b3, acc3);
    }

    const float* bias = (mat == 0) ? bq : ((mat == 1) ? bk : bv);
    v8f* accs[4] = {&acc0, &acc1, &acc2, &acc3};

#pragma unroll
    for (int t = 0; t < 4; ++t) {
        const int n0 = col0 + t * 16;
        const float bcol = bias[n0 + m];
        v8f& acc = *accs[t];
        if (mat < 2) {
            __bf16* dst = (mat == 0) ? Qb : Kb;
#pragma unroll
            for (int r = 0; r < 8; ++r) {
                int row = (lane < 16) ? r : (r + 8);
                dst[(size_t)(rowTile * 16 + row) * Hn + n0 + m] = (__bf16)(acc[r] + bcol);
            }
        } else {
#pragma unroll
            for (int r = 0; r < 8; ++r) {
                int row   = (lane < 16) ? r : (r + 8);
                int srow  = rowTile * 16 + row;      // global index over B*S
                int batch = srow >> 11;
                int s     = srow & (Sn - 1);
                Vt[(size_t)batch * Hn * Sn + (size_t)(n0 + m) * Sn + s] = (__bf16)(acc[r] + bcol);
            }
        }
    }
}

// ---------------- fused attention (scores + softmax + P@V) ----------------
// One workgroup (8 waves, 256 threads) per (batch, 16-query tile).
// LDS: 16x2048 f32 scores (128KB) + 16x2048 bf16 probs (64KB) + 16x512 bf16 Q (16KB).
__global__ __launch_bounds__(256) void attn_kernel(const __bf16* __restrict__ Qb,
                                                   const __bf16* __restrict__ Kb,
                                                   const __bf16* __restrict__ Vt,
                                                   float* __restrict__ out) {
    extern __shared__ char smem[];
    float*  sc = (float*)smem;                               // [16][2048] f32
    __bf16* pb = (__bf16*)(smem + 16 * Sn * 4);              // [16][2048] bf16 probs
    __bf16* qs = (__bf16*)(smem + 16 * Sn * 4 + 16 * Sn * 2);// [16][512]  bf16 Q

    const int batch = blockIdx.x >> 7;   // 128 q-tiles per batch
    const int q0    = (blockIdx.x & 127) << 4;
    const int tid   = threadIdx.x;
    const int lane  = tid & 31;
    const int wv    = tid >> 5;          // wave 0..7

    // ---- stage Q tile into LDS ----
    const __bf16* qsrc = Qb + ((size_t)batch * Sn + q0) * Hn;
    for (int i = tid; i < 16 * Hn / 8; i += 256)
        ((v8bf*)qs)[i] = ((const v8bf*)qsrc)[i];
    __syncthreads();

    const int m  = lane & 15;
    const int kb = (lane < 16) ? 0 : 8;

    // ---- phase 1: scores, 256 keys per wave, 2 key tiles share each Q-frag ----
    const __bf16* qrow = qs + m * Hn;
#pragma unroll 1
    for (int t = 0; t < 8; ++t) {
        const int key0 = wv * 256 + t * 32;
        const __bf16* krow0 = Kb + ((size_t)batch * Sn + key0 +  0 + m) * Hn;
        const __bf16* krow1 = Kb + ((size_t)batch * Sn + key0 + 16 + m) * Hn;
        v8f acc0 = {}, acc1 = {};
#pragma unroll 2
        for (int k = 0; k < Hn; k += 32) {
            v16bf a  = make_frag(qrow  + k + kb, qrow  + k + 16 + kb);
            v16bf b0 = make_frag(krow0 + k + kb, krow0 + k + 16 + kb);
            v16bf b1 = make_frag(krow1 + k + kb, krow1 + k + 16 + kb);
            acc0 = wmma_bf16(a, b0, acc0);
            acc1 = wmma_bf16(a, b1, acc1);
        }
#pragma unroll
        for (int r = 0; r < 8; ++r) {
            int row = (lane < 16) ? r : (r + 8);
            sc[row * Sn + key0 + m]      = acc0[r];
            sc[row * Sn + key0 + 16 + m] = acc1[r];
        }
    }
    __syncthreads();

    // ---- phase 2: softmax (16 lanes per row; NO 1/sqrt(d) scaling),
    //      normalized probs written once as bf16 ----
    {
        const int row = tid >> 4;        // 0..15
        const int j   = tid & 15;
        float*  srow = sc + row * Sn;
        __bf16* prow = pb + row * Sn;
        float mx = -3.402823466e38f;
        for (int c = j; c < Sn; c += 16) mx = fmaxf(mx, srow[c]);
#pragma unroll
        for (int msk = 8; msk >= 1; msk >>= 1) mx = fmaxf(mx, __shfl_xor(mx, msk, 32));
        float sum = 0.f;
        for (int c = j; c < Sn; c += 16) {
            float e = __expf(srow[c] - mx);
            srow[c] = e;
            sum += e;
        }
#pragma unroll
        for (int msk = 8; msk >= 1; msk >>= 1) sum += __shfl_xor(sum, msk, 32);
        const float inv = 1.0f / sum;
        for (int c = j; c < Sn; c += 16) prow[c] = (__bf16)(srow[c] * inv);
    }
    __syncthreads();

    // ---- phase 3: out[16, h0:h0+64] = probs @ V; probs A-frag (LDS bf16)
    //      shared across 4 H-tiles ----
    const int h0 = wv * 64;
    const __bf16* vrow0 = Vt + ((size_t)batch * Hn + h0 +  0 + m) * Sn;
    const __bf16* vrow1 = Vt + ((size_t)batch * Hn + h0 + 16 + m) * Sn;
    const __bf16* vrow2 = Vt + ((size_t)batch * Hn + h0 + 32 + m) * Sn;
    const __bf16* vrow3 = Vt + ((size_t)batch * Hn + h0 + 48 + m) * Sn;
    const __bf16* prow  = pb + m * Sn;

    v8f acc0 = {}, acc1 = {}, acc2 = {}, acc3 = {};
#pragma unroll 2
    for (int kk = 0; kk < Sn; kk += 32) {
        v16bf a  = make_frag(prow  + kk + kb, prow  + kk + 16 + kb);
        v16bf b0 = make_frag(vrow0 + kk + kb, vrow0 + kk + 16 + kb);
        v16bf b1 = make_frag(vrow1 + kk + kb, vrow1 + kk + 16 + kb);
        v16bf b2 = make_frag(vrow2 + kk + kb, vrow2 + kk + 16 + kb);
        v16bf b3 = make_frag(vrow3 + kk + kb, vrow3 + kk + 16 + kb);
        acc0 = wmma_bf16(a, b0, acc0);
        acc1 = wmma_bf16(a, b1, acc1);
        acc2 = wmma_bf16(a, b2, acc2);
        acc3 = wmma_bf16(a, b3, acc3);
    }

    v8f* accs[4] = {&acc0, &acc1, &acc2, &acc3};
#pragma unroll
    for (int t = 0; t < 4; ++t) {
        v8f& acc = *accs[t];
#pragma unroll
        for (int r = 0; r < 8; ++r) {
            int row = (lane < 16) ? r : (r + 8);
            out[((size_t)batch * Sn + q0 + row) * Hn + h0 + t * 16 + m] = acc[r];
        }
    }
}

// ---------------- launch ----------------

extern "C" void kernel_launch(void* const* d_in, const int* in_sizes, int n_in,
                              void* d_out, int out_size, void* d_ws, size_t ws_size,
                              hipStream_t stream) {
    const float* x  = (const float*)d_in[0];
    const float* Wq = (const float*)d_in[1];
    const float* bq = (const float*)d_in[2];
    const float* Wk = (const float*)d_in[3];
    const float* bk = (const float*)d_in[4];
    const float* Wv = (const float*)d_in[5];
    const float* bv = (const float*)d_in[6];
    float* out = (float*)d_out;

    // workspace layout (bf16 buffers), ~68.7 MB total
    char* ws = (char*)d_ws;
    size_t off = 0;
    __bf16* xb = (__bf16*)(ws + off); off += (size_t)Bn * Sn * Dn * 2;   // 16 MB
    __bf16* Wt = (__bf16*)(ws + off); off += (size_t)3 * Dn * Hn * 2;    // 1.5 MB
    __bf16* Qb = (__bf16*)(ws + off); off += (size_t)Bn * Sn * Hn * 2;   // 16 MB
    __bf16* Kb = (__bf16*)(ws + off); off += (size_t)Bn * Sn * Hn * 2;   // 16 MB
    __bf16* Vt = (__bf16*)(ws + off); off += (size_t)Bn * Hn * Sn * 2;   // 16 MB
    (void)ws_size; (void)out_size; (void)n_in; (void)in_sizes;

    // 1) precision conversion / layout transforms
    const int nx = Bn * Sn * Dn;
    cvt_x_kernel<<<2048, 256, 0, stream>>>(x, xb, nx);
    cvt_w_kernel<<<(3 * Dn * Hn) / 256, 256, 0, stream>>>(Wq, Wk, Wv, Wt);

    // 2) QKV projection: 3 * 1024 * 8 tile-groups, 4 waves per block
    qkv_kernel<<<(3 * 1024 * 8) / 4, 128, 0, stream>>>(xb, Wt, bq, bk, bv, Qb, Kb, Vt);

    // 3) fused attention: one block per (batch, 16-query tile)
    const int shmem = 16 * Sn * 4 + 16 * Sn * 2 + 16 * Hn * 2;  // 128K + 64K + 16K = 208KB
    (void)hipFuncSetAttribute((const void*)attn_kernel,
                              hipFuncAttributeMaxDynamicSharedMemorySize, shmem);
    attn_kernel<<<Bn * (Sn / 16), 256, shmem, stream>>>(Qb, Kb, Vt, out);
}
